// Encoder_33432025432390
// MI455X (gfx1250) — compile-verified, matrix operand order checked by reading, not run on hardware
//
#include <hip/hip_runtime.h>
#include <math.h>

// ---------------------------------------------------------------------------
// MI455X (gfx1250) LayerNorm-GRU encoder, v3.
// - v_wmma_f32_16x16x32_bf16 everywhere, f32 accumulate
// - weights pre-swizzled to fragment-ready tiles (1 x 32B load per B fragment)
// - A tiles staged in LDS via global_load_async_to_lds_b128 (+ s_wait_asynccnt)
// ---------------------------------------------------------------------------

typedef __bf16 bf16_t;
typedef __attribute__((ext_vector_type(16))) __bf16 v16bf;
typedef __attribute__((ext_vector_type(8)))  float  v8f;
typedef __attribute__((ext_vector_type(4)))  int    v4i;

#define WMMA_BF16(a, b, c) \
  __builtin_amdgcn_wmma_f32_16x16x32_bf16(false, (a), false, (b), (short)0, (c), false, false)

constexpr int  Bc = 64, Sc = 256, Ec = 1024, Hc = 512, G3 = 1536;
constexpr int  Mrows = Bc * Sc;            // 16384
constexpr float EPS = 1e-5f;

// K index held by (lane, element e) for 16-bit 16x32 A/B fragments (ISA 7.12.2).
__device__ __forceinline__ int frag_k(int lane, int e) {
  int v = e >> 1;
  return ((v >> 2) << 4) + ((lane >> 4) << 3) + ((v & 3) << 1) + (e & 1);
}

// A fragment (16x32) gathered from bf16 row-major tile (2 contiguous 16B runs/lane).
__device__ __forceinline__ v16bf load_a_bf(const bf16_t* sm, int ld) {
  int lane = threadIdx.x & 31;
  int m = lane & 15;
  v16bf f;
#pragma unroll
  for (int e = 0; e < 16; ++e) f[e] = sm[m * ld + frag_k(lane, e)];
  return f;
}

// B fragment from a pre-swizzled tile: lane-major, 16 contiguous bf16 per lane.
__device__ __forceinline__ v16bf load_b_packed(const bf16_t* tile) {
  return *(const v16bf*)(tile + (size_t)(threadIdx.x & 31) * 16);
}

__device__ __forceinline__ float sigmoidf_(float x) { return 1.f / (1.f + __expf(-x)); }

// 16B global -> LDS copy; async path when the gfx1250 builtin is available.
__device__ __forceinline__ void copy16_g2l(bf16_t* ldst, const bf16_t* gsrc) {
#if defined(__gfx1250__) && __has_builtin(__builtin_amdgcn_global_load_async_to_lds_b128)
  __builtin_amdgcn_global_load_async_to_lds_b128((v4i*)gsrc, (v4i*)ldst, 0, 0);
#else
  *(float4*)ldst = *(const float4*)gsrc;
#endif
}
__device__ __forceinline__ void async_fence_() {
#if defined(__gfx1250__) && __has_builtin(__builtin_amdgcn_global_load_async_to_lds_b128)
#if __has_builtin(__builtin_amdgcn_s_wait_asynccnt)
  __builtin_amdgcn_s_wait_asynccnt(0);
#else
  asm volatile("s_wait_asynccnt 0" ::: "memory");
#endif
#endif
}

// ---------------------------------------------------------------------------
// f32 -> bf16 elementwise convert
// ---------------------------------------------------------------------------
__global__ void cvt_f32_bf16(const float* __restrict__ s, bf16_t* __restrict__ d, int n) {
  int i = blockIdx.x * 1024 + threadIdx.x;
#pragma unroll
  for (int k = 0; k < 4; ++k) {
    int j = i + k * 256;
    if (j < n) d[j] = (bf16_t)s[j];
  }
}

// gather h0 = x[:, -1]
__global__ void gather_last(const float* __restrict__ outs, float* __restrict__ h) {
  int i = blockIdx.x * 256 + threadIdx.x;      // 65536 total
  int b = i >> 10, c = i & 1023;
  h[i] = outs[(size_t)(b * Sc + (Sc - 1)) * 1024 + c];
}

// ---------------------------------------------------------------------------
// Pack W (KxN f32 row-major) into WMMA-fragment-ready bf16 tiles:
//   tile = nt*(K/32)+kt ; out[(tile*32 + lane)*16 + e] = W[(kt*32+frag_k)*N + nt*16 + lane%16]
// One wave per tile; 8 tiles per 256-thread block.
// ---------------------------------------------------------------------------
__global__ __launch_bounds__(256)
void pack_b_frag(const float* __restrict__ W, bf16_t* __restrict__ out, int K, int N) {
  int tile = blockIdx.x * 8 + (threadIdx.x >> 5);
  int lane = threadIdx.x & 31;
  int numKt = K >> 5;
  int nt = tile / numKt, kt = tile - nt * numKt;
  const float* src = W + (size_t)(kt * 32) * N + nt * 16 + (lane & 15);
  bf16_t tmp[16];
#pragma unroll
  for (int e = 0; e < 16; ++e) tmp[e] = (bf16_t)src[(size_t)frag_k(lane, e) * N];
  bf16_t* dst = out + ((size_t)tile * 32 + lane) * 16;
  *(float4*)(dst)     = *(const float4*)(tmp);
  *(float4*)(dst + 8) = *(const float4*)(tmp + 8);
}

// ---------------------------------------------------------------------------
// xg = LN_gate(x @ Wx + bx) * gx + bxl -> bf16
// Block: 16 rows x 512 cols (one gate); A row-block (16x1024) staged once.
// grid = (M/16, 3 gates, 2 dirs)
// ---------------------------------------------------------------------------
__global__ __launch_bounds__(256)
void xg_gemm_ln(const bf16_t* __restrict__ Xb, const bf16_t* __restrict__ Wxp,
                const float* __restrict__ bx, const float* __restrict__ gx,
                const float* __restrict__ bxl, bf16_t* __restrict__ xg) {
  __shared__ bf16_t As[16 * 1024];     // 32 KB
  __shared__ float sred[16], qred[16];

  const int tid = threadIdx.x, wave = tid >> 5, lane = tid & 31;
  const int half = lane >> 4, n = lane & 15;
  const int row0 = blockIdx.x * 16;
  const int gate = blockIdx.y;
  const int dir  = blockIdx.z;

  const bf16_t* Wp  = Wxp + (size_t)dir * Ec * G3;   // packed: (G3/16 tiles) x (Ec/32) x 512
  const float*  bxd = bx  + dir * G3 + gate * 512;
  const float*  gxd = gx  + dir * G3 + gate * 512;
  const float*  bld = bxl + dir * G3 + gate * 512;
  bf16_t*       out = xg  + (size_t)dir * (size_t)Mrows * G3;

  if (tid < 16) { sred[tid] = 0.f; qred[tid] = 0.f; }
  // stage A block: 16 rows x 1024 cols = 2048 x 16B chunks
  for (int i = tid; i < 2048; i += 256) {
    int r = i >> 7, c = (i & 127) << 3;
    copy16_g2l(&As[r * 1024 + c], &Xb[(size_t)(row0 + r) * Ec + c]);
  }
  async_fence_();
  __syncthreads();

  v8f acc[4] = {};
  const int nt0 = gate * 32 + wave * 4;              // global 16-col tile index
  for (int kt = 0; kt < 32; ++kt) {
    v16bf a = load_a_bf(&As[kt * 32], 1024);
#pragma unroll
    for (int t2 = 0; t2 < 4; ++t2) {
      v16bf b = load_b_packed(Wp + ((size_t)(nt0 + t2) * 32 + kt) * 512);
      acc[t2] = WMMA_BF16(a, b, acc[t2]);
    }
  }

  // bias + LN stats (row = r + 8*half, col = wave*64 + t2*16 + n)
  float y[4][8];
#pragma unroll
  for (int t2 = 0; t2 < 4; ++t2) {
    int col = wave * 64 + t2 * 16 + n;
#pragma unroll
    for (int r = 0; r < 8; ++r) y[t2][r] = acc[t2][r] + bxd[col];
  }
#pragma unroll
  for (int r = 0; r < 8; ++r) {
    float s = 0.f, q = 0.f;
#pragma unroll
    for (int t2 = 0; t2 < 4; ++t2) { float v = y[t2][r]; s += v; q += v * v; }
    int row = r + (half << 3);
    atomicAdd(&sred[row], s);
    atomicAdd(&qred[row], q);
  }
  __syncthreads();
#pragma unroll
  for (int r = 0; r < 8; ++r) {
    int row = r + (half << 3);
    float mu = sred[row] * (1.f / 512.f);
    float var = qred[row] * (1.f / 512.f) - mu * mu;
    float rs = rsqrtf(var + EPS);
#pragma unroll
    for (int t2 = 0; t2 < 4; ++t2) {
      int col = wave * 64 + t2 * 16 + n;
      float v = (y[t2][r] - mu) * rs * gxd[col] + bld[col];
      out[(size_t)(row0 + row) * G3 + gate * 512 + col] = (bf16_t)v;
    }
  }
}

// ---------------------------------------------------------------------------
// Sequential GRU scan. One WG per (batch-tile of 16, dir).
// h kept in LDS as f32 master + bf16 mirror; Wh streamed packed from L2.
// grid = (4, 2)
// ---------------------------------------------------------------------------
__global__ __launch_bounds__(256)
void gru_scan(const bf16_t* __restrict__ xg, const bf16_t* __restrict__ whp,
              const float* __restrict__ bh, const float* __restrict__ gh,
              const float* __restrict__ bhl, float* __restrict__ outs) {
  __shared__ float  hS[16 * 512];        // 32 KB f32 h state
  __shared__ bf16_t hbS[16 * 512];       // 16 KB bf16 mirror (WMMA A)
  __shared__ bf16_t hgS[16 * 1536];      // 48 KB hidden-gate pre-activations
  __shared__ float  sred[16 * 3], qred[16 * 3];

  const int tid = threadIdx.x, wave = tid >> 5, lane = tid & 31;
  const int half = lane >> 4, n = lane & 15;
  const int b0 = blockIdx.x * 16;
  const int dir = blockIdx.y;

  const bf16_t* Wp  = whp + (size_t)dir * Hc * G3;   // packed: 96 x 16 x 512
  const bf16_t* xgd = xg  + (size_t)dir * (size_t)Mrows * G3;
  const float*  bhd = bh  + dir * G3;
  const float*  ghd = gh  + dir * G3;
  const float*  bld = bhl + dir * G3;

  for (int i = tid; i < 16 * 512; i += 256) { hS[i] = 0.f; hbS[i] = (bf16_t)0.f; }
  __syncthreads();

  for (int t = 0; t < Sc; ++t) {
    if (tid < 48) { sred[tid] = 0.f; qred[tid] = 0.f; }
    __syncthreads();

    // hg = h @ Wh + bh : each wave owns 12 N-tiles of 16 cols
    for (int nt = 0; nt < 12; ++nt) {
      const bf16_t* bb = Wp + ((size_t)(wave * 12 + nt) * 16) * 512;
      v8f acc = {};
      for (int kt = 0; kt < 16; ++kt) {
        __builtin_prefetch(bb + (kt + 2) * 512, 0, 1);
        v16bf a = load_a_bf(&hbS[kt * 32], 512);
        v16bf b = load_b_packed(bb + (size_t)kt * 512);
        acc = WMMA_BF16(a, b, acc);
      }
      int col = wave * 192 + nt * 16 + n;
      int gate = col >> 9;
#pragma unroll
      for (int r = 0; r < 8; ++r) {
        int row = r + (half << 3);
        float v = acc[r] + bhd[col];
        hgS[row * 1536 + col] = (bf16_t)v;
        atomicAdd(&sred[row * 3 + gate], v);
        atomicAdd(&qred[row * 3 + gate], v * v);
      }
    }
    __syncthreads();

    // fused LN + gate math + state update
    for (int i = tid; i < 16 * 512; i += 256) {
      int row = i >> 9, j = i & 511;
      int m = (b0 + row) * Sc + t;
      float mur = sred[row * 3 + 0] * (1.f / 512.f);
      float muz = sred[row * 3 + 1] * (1.f / 512.f);
      float mun = sred[row * 3 + 2] * (1.f / 512.f);
      float rsr = rsqrtf(qred[row * 3 + 0] * (1.f / 512.f) - mur * mur + EPS);
      float rsz = rsqrtf(qred[row * 3 + 1] * (1.f / 512.f) - muz * muz + EPS);
      float rsn = rsqrtf(qred[row * 3 + 2] * (1.f / 512.f) - mun * mun + EPS);
      float hr = ((float)hgS[row * 1536 + j]        - mur) * rsr * ghd[j]        + bld[j];
      float hz = ((float)hgS[row * 1536 + 512 + j]  - muz) * rsz * ghd[512 + j]  + bld[512 + j];
      float hn = ((float)hgS[row * 1536 + 1024 + j] - mun) * rsn * ghd[1024 + j] + bld[1024 + j];
      float xr = (float)xgd[(size_t)m * G3 + j];
      float xz = (float)xgd[(size_t)m * G3 + 512 + j];
      float xn = (float)xgd[(size_t)m * G3 + 1024 + j];
      float rg = sigmoidf_(xr + hr);
      float zg = sigmoidf_(xz + hz);
      float ng = tanhf(xn + rg * hn);
      float hold = hS[i];
      float hnew = (1.f - zg) * ng + zg * hold;
      hS[i] = hnew;
      hbS[i] = (bf16_t)hnew;
      outs[(size_t)m * 1024 + dir * 512 + j] = hnew;
    }
    __syncthreads();
  }
}

// ---------------------------------------------------------------------------
// Highway layer: g = sigmoid(h@Wg+bg); h' = g*elu(h@Wh+bh) + (1-g)*h
// 8 waves: 0-3 gate GEMM, 4-7 transform GEMM; tile 16 rows x 64 cols.
// grid = (4, 16)
// ---------------------------------------------------------------------------
__global__ __launch_bounds__(256)
void highway(const bf16_t* __restrict__ hb, const bf16_t* __restrict__ Wgp,
             const bf16_t* __restrict__ Whp, const float* __restrict__ bg,
             const float* __restrict__ bhw, const float* __restrict__ hin,
             float* __restrict__ hout, bf16_t* __restrict__ hbout) {
  __shared__ bf16_t As[16 * 1024];       // 32 KB
  __shared__ float  tmp[2][16][64];
  const int tid = threadIdx.x, wave = tid >> 5, lane = tid & 31;
  const int half = lane >> 4, n = lane & 15;
  const int r0 = blockIdx.x * 16;
  const int c0 = blockIdx.y * 64;
  const int sel = wave >> 2;
  const int ct = wave & 3;
  const bf16_t* Wp = sel ? Whp : Wgp;    // packed: 64 x 32 x 512

  for (int i = tid; i < 2048; i += 256) {
    int r = i >> 7, c = (i & 127) << 3;
    copy16_g2l(&As[r * 1024 + c], &hb[(size_t)(r0 + r) * 1024 + c]);
  }
  async_fence_();
  __syncthreads();

  v8f acc = {};
  const int ntg = (c0 >> 4) + ct;
  for (int kt = 0; kt < 32; ++kt) {
    v16bf a = load_a_bf(&As[kt * 32], 1024);
    v16bf b = load_b_packed(Wp + ((size_t)ntg * 32 + kt) * 512);
    acc = WMMA_BF16(a, b, acc);
  }
  const float* bias = sel ? bhw : bg;
#pragma unroll
  for (int r = 0; r < 8; ++r) {
    int row = r + (half << 3), col = ct * 16 + n;
    tmp[sel][row][col] = acc[r] + bias[c0 + col];
  }
  __syncthreads();
  for (int i = tid; i < 1024; i += 256) {
    int row = i >> 6, col = i & 63;
    float g = sigmoidf_(tmp[0][row][col]);
    float u = tmp[1][row][col];
    u = u > 0.f ? u : expm1f(u);
    float hold = hin[(size_t)(r0 + row) * 1024 + c0 + col];
    float v = g * u + (1.f - g) * hold;
    hout[(size_t)(r0 + row) * 1024 + c0 + col] = v;
    hbout[(size_t)(r0 + row) * 1024 + c0 + col] = (bf16_t)v;
  }
}

// ---------------------------------------------------------------------------
extern "C" void kernel_launch(void* const* d_in, const int* in_sizes, int n_in,
                              void* d_out, int out_size, void* d_ws, size_t ws_size,
                              hipStream_t stream) {
  const float* in_x = (const float*)d_in[0];
  const float* Wx   = (const float*)d_in[1];
  const float* Wh   = (const float*)d_in[2];
  const float* bx   = (const float*)d_in[3];
  const float* bhp  = (const float*)d_in[4];
  const float* gx   = (const float*)d_in[5];
  const float* bxl  = (const float*)d_in[6];
  const float* gh   = (const float*)d_in[7];
  const float* bhl  = (const float*)d_in[8];
  const float* hwWg = (const float*)d_in[9];
  const float* hwbg = (const float*)d_in[10];
  const float* hwWh = (const float*)d_in[11];
  const float* hwbh = (const float*)d_in[12];
  float* out = (float*)d_out;

  char* ws = (char*)d_ws;
  size_t off = 0;
  auto carve = [&](size_t bytes) -> void* {
    void* p = ws + off;
    off = (off + bytes + 255) & ~(size_t)255;
    return p;
  };
  bf16_t* xb   = (bf16_t*)carve((size_t)Mrows * 1024 * 2);       // layer input, bf16
  bf16_t* wxp  = (bf16_t*)carve((size_t)2 * Ec * G3 * 2);        // packed Wx[l]
  bf16_t* whp  = (bf16_t*)carve((size_t)2 * Hc * G3 * 2);        // packed Wh[l]
  bf16_t* xgb  = (bf16_t*)carve((size_t)2 * Mrows * G3 * 2);     // LN'd input gates
  float*  outs = (float*)carve((size_t)Mrows * 1024 * 4);        // layer output f32
  bf16_t* wgp  = (bf16_t*)carve((size_t)4 * 1024 * 1024 * 2);    // packed highway Wg
  bf16_t* whwp = (bf16_t*)carve((size_t)4 * 1024 * 1024 * 2);    // packed highway Wh
  float*  h1   = (float*)carve((size_t)64 * 1024 * 4);
  float*  h2   = (float*)carve((size_t)64 * 1024 * 4);
  bf16_t* hb1  = (bf16_t*)carve((size_t)64 * 1024 * 2);
  bf16_t* hb2  = (bf16_t*)carve((size_t)64 * 1024 * 2);

  auto cvt = [&](const float* s, bf16_t* d, size_t n) {
    cvt_f32_bf16<<<dim3((unsigned)((n + 1023) / 1024)), 256, 0, stream>>>(s, d, (int)n);
  };
  auto pack = [&](const float* W, bf16_t* P, int K, int N) {
    int tiles = (K / 32) * (N / 16);
    pack_b_frag<<<dim3(tiles / 8), 256, 0, stream>>>(W, P, K, N);
  };

  cvt(in_x, xb, (size_t)Mrows * 1024);
  for (int i = 0; i < 4; ++i) {
    pack(hwWg + (size_t)i * 1024 * 1024, wgp  + (size_t)i * 1024 * 1024, 1024, 1024);
    pack(hwWh + (size_t)i * 1024 * 1024, whwp + (size_t)i * 1024 * 1024, 1024, 1024);
  }

  for (int l = 0; l < 2; ++l) {
    for (int d = 0; d < 2; ++d) {
      pack(Wx + ((size_t)l * 2 + d) * Ec * G3, wxp + (size_t)d * Ec * G3, Ec, G3);
      pack(Wh + ((size_t)l * 2 + d) * Hc * G3, whp + (size_t)d * Hc * G3, Hc, G3);
    }
    xg_gemm_ln<<<dim3(Mrows / 16, 3, 2), 256, 0, stream>>>(
        xb, wxp, bx + l * 2 * G3, gx + l * 2 * G3, bxl + l * 2 * G3, xgb);
    gru_scan<<<dim3(4, 2), 256, 0, stream>>>(
        xgb, whp, bhp + l * 2 * G3, gh + l * 2 * G3, bhl + l * 2 * G3, outs);
    cvt(outs, xb, (size_t)Mrows * 1024);   // next layer's bf16 input
  }

  gather_last<<<dim3(256), 256, 0, stream>>>(outs, h1);
  cvt(h1, hb1, (size_t)64 * 1024);

  float*  hc = h1;  float*  hn = h2;
  bf16_t* hbc = hb1; bf16_t* hbn = hb2;
  for (int i = 0; i < 4; ++i) {
    float* dst = (i == 3) ? out : hn;
    highway<<<dim3(4, 16), 256, 0, stream>>>(
        hbc, wgp + (size_t)i * 1024 * 1024, whwp + (size_t)i * 1024 * 1024,
        hwbg + i * 1024, hwbh + i * 1024, hc, dst, hbn);
    float* tf = hc; hc = dst; hn = tf;
    bf16_t* tb = hbc; hbc = hbn; hbn = tb;
  }
}
